// LMInfinite_63342177681934
// MI455X (gfx1250) — matrix operand order repaired and checked by reading, not compiled, not run on hardware
//
#include <hip/hip_runtime.h>
#include <hip/hip_bf16.h>

#define SEQ        8192
#define DMODEL     128
#define NGLOBAL    100
#define LPRETRAIN  2048
#define NEG_BIG    (-1.0e30f)

typedef __attribute__((ext_vector_type(16))) _Float16 v16h;
typedef __attribute__((ext_vector_type(8)))  _Float16 v8h;
typedef __attribute__((ext_vector_type(4)))  _Float16 v4h;
typedef __attribute__((ext_vector_type(8)))  float    v8f;
typedef __attribute__((ext_vector_type(4)))  float    f32x4;

__device__ __forceinline__ float grp16_max(float x) {
  x = fmaxf(x, __shfl_xor(x, 1, 16));
  x = fmaxf(x, __shfl_xor(x, 2, 16));
  x = fmaxf(x, __shfl_xor(x, 4, 16));
  x = fmaxf(x, __shfl_xor(x, 8, 16));
  return x;
}
__device__ __forceinline__ float grp16_sum(float x) {
  x += __shfl_xor(x, 1, 16);
  x += __shfl_xor(x, 2, 16);
  x += __shfl_xor(x, 4, 16);
  x += __shfl_xor(x, 8, 16);
  return x;
}

// Block = 4 waves = 4 consecutive 16-query tiles (64 rows). The block walks a
// uniform union schedule of 32-key blocks; K/V are cooperatively loaded,
// converted to f16 ONCE, and staged in LDS (V transposed) for all 4 waves.
// Per wave: flash attention with v_wmma_f32_16x16x32_f16 for QK^T and PV.
// B-tiles are batch-loaded from LDS ahead of the WMMA runs so DS latency is
// pipelined (incremental s_wait_dscnt) instead of serialized per WMMA.
__global__ __launch_bounds__(128)
void lminfinite_attn(const float* __restrict__ Q, const float* __restrict__ K,
                     const float* __restrict__ V, float* __restrict__ O) {
  // K block: 32 keys x 128 dims, row pitch 136 halves (272B = 17*16B -> aligned,
  //          conflict-free b128 reads). 8704 B
  // Vt block: transposed, 128 dims x 32 keys, row pitch 40 halves (80B). 10240 B
  // P staging: per-wave 16x32 f16, pitch 40 halves. 5120 B
  alignas(16) __shared__ _Float16 ldsK[32 * 136];
  alignas(16) __shared__ _Float16 ldsVt[128 * 40];
  alignas(16) __shared__ _Float16 ldsP[4][16 * 40];

  const int tid  = threadIdx.x;
  const int lane = tid & 31;
  const int wave = tid >> 5;
  const int n    = lane & 15;   // column / lane-in-group
  const int hi   = lane >> 4;   // half-group select

  const int b     = blockIdx.x >> 7;            // 128 blocks per batch
  const int q0blk = (blockIdx.x & 127) * 64;    // first query row of block
  const int qw    = q0blk + wave * 16;          // this wave's query tile

  const float* Qb = Q + (size_t)b * SEQ * DMODEL;
  const float* Kb = K + (size_t)b * SEQ * DMODEL;
  const float* Vb = V + (size_t)b * SEQ * DMODEL;
  float*       Ob = O + (size_t)b * SEQ * DMODEL;

  // ---- Q tile (16x128) -> WMMA A layout (f16), 4 chunks of K=32.
  // A layout (16-bit, 16x32): lane row = n; elems 0..7 = cols base+8*hi+0..7,
  // elems 8..15 = cols base+16+8*hi+0..7.
  v16h aQ[4];
  {
    const float* qr = Qb + (size_t)(qw + n) * DMODEL;
#pragma unroll
    for (int c = 0; c < 4; ++c) {
      const float* p0 = qr + c * 32 + 8 * hi;
      f32x4 f0 = *(const f32x4*)(p0);
      f32x4 f1 = *(const f32x4*)(p0 + 4);
      f32x4 f2 = *(const f32x4*)(p0 + 16);
      f32x4 f3 = *(const f32x4*)(p0 + 20);
#pragma unroll
      for (int e = 0; e < 4; ++e) {
        aQ[c][e]      = (_Float16)f0[e];
        aQ[c][4 + e]  = (_Float16)f1[e];
        aQ[c][8 + e]  = (_Float16)f2[e];
        aQ[c][12 + e] = (_Float16)f3[e];
      }
    }
  }

  const v8f zero8 = {};
  v8f oAcc[8];   // C/D layout: elem v -> row v+8*hi, col 16*c+n
#pragma unroll
  for (int c = 0; c < 8; ++c) oAcc[c] = zero8;
  float mRow[8], lRow[8];
#pragma unroll
  for (int v = 0; v < 8; ++v) { mRow[v] = NEG_BIG; lRow[v] = 0.0f; }

  const float scale = 0.08838834764831845f; // 1/sqrt(128)
  _Float16* lp = ldsP[wave];

  // ---- block-uniform union key schedule
  const int jmaxU = q0blk + 63;
  int ls = q0blk - (LPRETRAIN - 1);
  if (ls < 0) ls = 0;
  ls &= ~31;
  const int start2 = (ls < 128) ? 128 : ls;     // global blocks are {0,32,64,96}

  auto advance = [&](int j0) -> int {
    int nj = j0 + 32;
    if (j0 < NGLOBAL) {                 // in global-sink range
      if (nj < NGLOBAL && nj <= jmaxU) return nj;
      nj = start2;                      // jump to local-window range
    }
    return (nj <= jmaxU) ? nj : -1;
  };

  // ---- cooperative staging pipeline registers (block i+1 loads hide under
  //      block i's WMMAs; s_wait lands at the next stage()'s converts)
  f32x4 kreg[8], vreg[8];
  auto prefetch = [&](int j0) {
#pragma unroll
    for (int k = 0; k < 8; ++k) {
      const int id  = tid + 128 * k;    // 0..1023 -> 32 rows x 32 col-groups
      const int row = id >> 5;
      const int cg  = id & 31;
      const size_t off = (size_t)(j0 + row) * DMODEL + cg * 4;
      kreg[k] = *(const f32x4*)(Kb + off);
      vreg[k] = *(const f32x4*)(Vb + off);
    }
  };
  auto stage = [&]() {
#pragma unroll
    for (int k = 0; k < 8; ++k) {
      const int id  = tid + 128 * k;
      const int row = id >> 5;
      const int cg  = id & 31;
      v4h hk;
#pragma unroll
      for (int e = 0; e < 4; ++e) hk[e] = (_Float16)kreg[k][e];
      *(v4h*)(ldsK + row * 136 + cg * 4) = hk;          // row-major K, shared cvt
#pragma unroll
      for (int e = 0; e < 4; ++e)                       // transposed V
        ldsVt[(cg * 4 + e) * 40 + row] = (_Float16)vreg[k][e];
    }
  };

  auto compute = [&](int j0) {
    // ---- batch-load ALL K B-tiles first: 16 ds_load_b128 in flight, waits
    // become incremental instead of s_wait_dscnt 0 per WMMA.
    v16h bk[8];
#pragma unroll
    for (int t = 0; t < 2; ++t)
#pragma unroll
      for (int c = 0; c < 4; ++c) {
        const _Float16* kr = ldsK + (16 * t + n) * 136 + c * 32 + 8 * hi;
        v8h k0 = *(const v8h*)(kr);
        v8h k1 = *(const v8h*)(kr + 16);
        v16h x;
#pragma unroll
        for (int e = 0; e < 8; ++e) { x[e] = k0[e]; x[8 + e] = k1[e]; }
        bk[t * 4 + c] = x;
      }

    // ---- S = Q K^T : two 16-key tiles, 4 chained WMMAs each (K=128)
    v8f s[2];
#pragma unroll
    for (int t = 0; t < 2; ++t) {
      v8f acc = zero8;
#pragma unroll
      for (int c = 0; c < 4; ++c)
        acc = __builtin_amdgcn_wmma_f32_16x16x32_f16(false, aQ[c], false,
                                                     bk[t * 4 + c], (short)0,
                                                     acc, false, false);
      s[t] = acc;
    }

    // ---- bias + lambda mask + online softmax. Fully-masked rows/blocks are
    // exact no-ops: key 0 (global sink) is always allowed, so mRow is finite
    // after the first block and exp(-1e30 - m) underflows to exactly 0.
    float alphaV[8];
#pragma unroll
    for (int v = 0; v < 8; ++v) {
      const int i = qw + v + 8 * hi;
      float mx = NEG_BIG;
#pragma unroll
      for (int t = 0; t < 2; ++t) {
        const int j = j0 + 16 * t + n;
        const bool ok = (j <= i) && ((j < NGLOBAL) || (j + (LPRETRAIN - 1) >= i));
        const float sv = ok ? (s[t][v] * scale + (float)(j - i)) : NEG_BIG;
        s[t][v] = sv;
        mx = fmaxf(mx, sv);
      }
      mx = grp16_max(mx);
      const float mNew  = fmaxf(mRow[v], mx);
      const float alpha = __expf(mRow[v] - mNew);
      float rs = 0.0f;
#pragma unroll
      for (int t = 0; t < 2; ++t) {
        const float p = __expf(s[t][v] - mNew);
        s[t][v] = p;
        rs += p;
      }
      lRow[v] = lRow[v] * alpha + grp16_sum(rs);
      mRow[v] = mNew;
      alphaV[v] = alpha;
    }

#pragma unroll
    for (int c = 0; c < 8; ++c)
#pragma unroll
      for (int v = 0; v < 8; ++v) oAcc[c][v] *= alphaV[v];

    // ---- batch-load ALL V B-tiles now, so they are in flight during the
    // P transpose round-trip below. B elem e<8 -> key 8*hi+e, e>=8 -> key 16+8*hi+(e-8), col c*16+n.
    v16h bv[8];
#pragma unroll
    for (int c = 0; c < 8; ++c) {
      const _Float16* vr = ldsVt + (c * 16 + n) * 40 + 8 * hi;
      v8h v0 = *(const v8h*)(vr);
      v8h v1 = *(const v8h*)(vr + 16);
      v16h x;
#pragma unroll
      for (int e = 0; e < 8; ++e) { x[e] = v0[e]; x[8 + e] = v1[e]; }
      bv[c] = x;
    }

    // ---- transpose P (C/D -> A layout) through per-wave LDS (in-order DS)
#pragma unroll
    for (int t = 0; t < 2; ++t)
#pragma unroll
      for (int v = 0; v < 8; ++v)
        lp[(v + 8 * hi) * 40 + 16 * t + n] = (_Float16)s[t][v];

    const _Float16* rp = lp + n * 40;
    v8h lo8 = *(const v8h*)(rp + 8 * hi);
    v8h hi8 = *(const v8h*)(rp + 16 + 8 * hi);
    v16h pA;
#pragma unroll
    for (int e = 0; e < 8; ++e) { pA[e] = lo8[e]; pA[8 + e] = hi8[e]; }

    // ---- O += P V : 8 independent WMMAs issue back-to-back
#pragma unroll
    for (int c = 0; c < 8; ++c)
      oAcc[c] = __builtin_amdgcn_wmma_f32_16x16x32_f16(false, pA, false, bv[c],
                                                       (short)0, oAcc[c],
                                                       false, false);
  };

  // ---- main pipelined loop (block-uniform control flow; barriers legal)
  int j0 = 0;
  prefetch(0);
  while (true) {
    stage();                // wait loads, convert once, ds-store shared tiles
    __syncthreads();
    const int jn = advance(j0);
    if (jn >= 0) prefetch(jn);   // next block's global loads hide under WMMAs
    compute(j0);
    __syncthreads();
    if (jn < 0) break;
    j0 = jn;
  }

  // ---- epilogue: normalize and store (coalesced 64B per 16-lane group)
#pragma unroll
  for (int v = 0; v < 8; ++v) {
    const float inv = 1.0f / lRow[v];
    float* orow = Ob + (size_t)(qw + v + 8 * hi) * DMODEL;
#pragma unroll
    for (int c = 0; c < 8; ++c) orow[c * 16 + n] = oAcc[c][v] * inv;
  }
}

extern "C" void kernel_launch(void* const* d_in, const int* in_sizes, int n_in,
                              void* d_out, int out_size, void* d_ws, size_t ws_size,
                              hipStream_t stream) {
  (void)in_sizes; (void)n_in; (void)out_size; (void)d_ws; (void)ws_size;
  const float* q = (const float*)d_in[0];
  const float* k = (const float*)d_in[1];
  const float* v = (const float*)d_in[2];
  float* o = (float*)d_out;

  const int B = 2;
  const int blocks = B * (SEQ / 64);    // 256 blocks, 4 waves each
  lminfinite_attn<<<dim3(blocks), dim3(128), 0, stream>>>(q, k, v, o);
}
